// BioacousticHORN_39256001085447
// MI455X (gfx1250) — matrix operand order repaired and checked by reading, not compiled, not run on hardware
//
#include <hip/hip_runtime.h>
#include <hip/hip_bf16.h>

typedef __attribute__((ext_vector_type(16))) __bf16 v16bf;
typedef __attribute__((ext_vector_type(8)))  float  v8f;

namespace {
constexpr int kB = 256, kS = 512, kD = 128, kR = 2048;
constexpr float kH = 0.5f;
constexpr int kYS = 2052;                      // padded LDS row stride (floats): 8*2052 % 64 == 32
// d_ws byte offsets
constexpr size_t kOffF1   = 0;                 // 512 KB  W_proj fragments (bf16)
constexpr size_t kOffF2   = 512*1024;          // 512 KB  W_read fragments (bf16)
constexpr size_t kOffPR   = 1024*1024;         // 32 KB   float4 res params [R]
constexpr size_t kOffPO   = kOffPR + 32*1024;  // 32 KB   float4 out params [R]
constexpr size_t kOffPIW  = kOffPO + 32*1024;  // 512 B   w2 input params [D]
constexpr size_t kOffPIG  = kOffPIW + 512;     // 512 B   2|gamma| input [D]
constexpr size_t kOffPIA  = kOffPIG + 512;     // 512 B   alpha input [D]
constexpr size_t kOffPIB  = kOffPIA + 512;     // 512 B   b_read [D]
constexpr size_t kOffPart = kOffPIB + 512;     // 128 B   per-WG loss partials
}

__device__ __forceinline__ float clamp5(float v){ return fminf(5.f, fmaxf(-5.f, v)); }

__device__ __forceinline__ v8f wmma_bf16(v16bf a, v16bf b, v8f c){
  return __builtin_amdgcn_wmma_f32_16x16x32_bf16(false, a, false, b, (short)0, c, false, false);
}

// 16-bit A-fragment scatter (used for horn_main's per-step A1 staging):
__device__ __forceinline__ int stage_a_idx(int m, int k){
  int kt = k >> 5, kk = k & 31;
  int L, j;
  if      (kk <  8) { L = m;      j = kk;      }
  else if (kk < 16) { L = m + 16; j = kk - 8;  }
  else if (kk < 24) { L = m;      j = kk - 8;  }
  else              { L = m + 16; j = kk - 16; }
  return (kt*32 + L)*16 + j;
}

// One-time prep: bf16 B-fragment tiles for both GEMMs + packed oscillator params.
__global__ void horn_prep(const float* __restrict__ Wp, const float* __restrict__ Wr,
                          const float* __restrict__ omega_in, const float* __restrict__ gamma_in,
                          const float* __restrict__ alpha_in, const float* __restrict__ b_proj,
                          const float* __restrict__ omega_res, const float* __restrict__ gamma_res,
                          const float* __restrict__ alpha_res, const float* __restrict__ omega_out,
                          const float* __restrict__ gamma_out, const float* __restrict__ alpha_out,
                          const float* __restrict__ b_read,
                          __bf16* __restrict__ f1, __bf16* __restrict__ f2,
                          float4* __restrict__ pr4, float4* __restrict__ po4,
                          float* __restrict__ piW, float* __restrict__ piG,
                          float* __restrict__ piA, float* __restrict__ piB){
  int idx = blockIdx.x*blockDim.x + threadIdx.x;   // 0..262143
  int j = idx & 15, L = (idx >> 4) & 31;
  int koff = ((L >> 4) << 3) + j + ((j >= 8) ? 8 : 0);
  int kt  = (idx >> 9) & 3,  nt  = idx >> 11;
  f1[idx] = (__bf16)Wp[(nt*16 + (L & 15))*kD + kt*32 + koff];
  int kt2 = (idx >> 9) & 63, nt2 = idx >> 15;
  f2[idx] = (__bf16)Wr[(nt2*16 + (L & 15))*kR + kt2*32 + koff];
  if (idx < kR){
    float om = omega_res[idx], oo = omega_out[idx];
    pr4[idx] = make_float4(om*om, 2.f*fabsf(gamma_res[idx]), alpha_res[idx], b_proj[idx]);
    po4[idx] = make_float4(oo*oo, 2.f*fabsf(gamma_out[idx]), alpha_out[idx], 0.f);
  }
  if (idx < kD){
    float oi = omega_in[idx];
    piW[idx] = oi*oi;
    piG[idx] = 2.f*fabsf(gamma_in[idx]);
    piA[idx] = alpha_in[idx];
    piB[idx] = b_read[idx];
  }
}

__global__ __launch_bounds__(512, 1) void horn_main(
    const float* __restrict__ inputs,
    const v16bf* __restrict__ gB1, const v16bf* __restrict__ gB2,
    const float4* __restrict__ pr4, const float4* __restrict__ po4,
    const float* __restrict__ piW, const float* __restrict__ piG,
    const float* __restrict__ piA, const float* __restrict__ piB,
    float* __restrict__ preds, float* __restrict__ partials)
{
  __shared__ __align__(32) __bf16 sA1[4*32*16];     // x_in  A-fragments (16x128)        4 KB
  __shared__ __align__(32) __bf16 sA2[32*32*16];    // x_out A-fragments (half K)       32 KB
  __shared__ float sPred[2048];                     // pred accumulator / reduce         8 KB
  __shared__ float sYr[16*kYS];                     // y_res state [m][r padded]     128.25 KB
  __shared__ float sYo[16*kYS];                     // y_out state [m][r padded]     128.25 KB

  const int tid  = threadIdx.x;                     // 0..511 (16 waves)
  const int w    = tid >> 5;                        // wave 0..15
  const int lane = tid & 31;
  const int hi8  = (lane >> 4) << 3;
  const int n0   = lane & 15;

  // ---- per-thread bases (single registers; all inner offsets are unroll constants) ----
  const int elem   = tid*4;                         // phase A/E element group
  const int mrow_a = elem >> 7;
  const int k0     = elem & 127;                    // multiple of 4
  const long gbase = ((long)(blockIdx.x*16 + mrow_a)*kS)*kD + k0;
  const int a1sIdx = stage_a_idx(mrow_a, k0);       // A1 staging: 4 consecutive bf16 here
  // A2 staging base: index = stBase + (tt>>1)*512 + (tt&1)*8 + v*16
  const int stBase = (w & 7)*2048 + (hi8 << 4) + ((n0 >= 8) ? 256 : 0) + (n0 & 7);
  const int yB     = hi8*kYS + w*128 + n0;          // y state base (floats)
  const int a1Base = lane;                          // v16bf units into sA1
  const int nt2    = w & 7;                         // readout N tile
  const int ktbase = (w >> 3)*16;                   // readout chunk sub-range per K half
  const int a2Base = ktbase*32 + lane;              // v16bf units into sA2
  const int pBase  = hi8*kD + nt2*16 + n0;          // sPred accumulate base

  float x[4] = {0.f,0.f,0.f,0.f}, y[4] = {0.f,0.f,0.f,0.f};
  const v8f zero8 = {0.f,0.f,0.f,0.f,0.f,0.f,0.f,0.f};
  v8f Xr[8], Xo[8];
#pragma unroll
  for (int tt=0;tt<8;++tt){ Xr[tt]=zero8; Xo[tt]=zero8; }

  float lin = 0.f, lout = 0.f;
  *(float4*)&sPred[elem] = make_float4(0.f,0.f,0.f,0.f);
  for (int i = tid; i < 16*kYS; i += 512){ sYr[i] = 0.f; sYo[i] = 0.f; }
  __syncthreads();

  const v16bf* a1v = (const v16bf*)sA1;
  const v16bf* a2v = (const v16bf*)sA2;

  for (int t = 0; t < kS; ++t){
    // Launder a zero offset each step: blocks LICM from hoisting the L2-resident
    // weight/param loads out of the time loop, without breaking addrspace inference.
    int zoff = 0;
    asm volatile("" : "+s"(zoff));
    const int b1Idx = w*1024 + lane + zoff;               // v16bf units into gB1
    const int b2Idx = (nt2*64 + ktbase)*32 + lane + zoff; // v16bf units into gB2
    const int prIdx = w*128 + n0 + zoff;                  // float4 units
    const int piIdx = k0 + zoff;

    // ---- phase A: input oscillator + pc_loss_in + packed A1 staging ----
    {
      const float4 inp = *(const float4*)&inputs[gbase + (long)t*kD];
      const float4 wi = *(const float4*)&piW[piIdx];
      const float4 gi = *(const float4*)&piG[piIdx];
      const float4 ai = *(const float4*)&piA[piIdx];
      const float in4[4] = {inp.x, inp.y, inp.z, inp.w};
      const float wi4[4] = {wi.x, wi.y, wi.z, wi.w};
      const float gi4[4] = {gi.x, gi.y, gi.z, gi.w};
      const float ai4[4] = {ai.x, ai.y, ai.z, ai.w};
      union { __bf16 h[4]; uint2 u; } pk;
#pragma unroll
      for (int e=0;e<4;++e){
        if (t){ float d = x[e] - in4[e]; lin += d*d; }
        float nx = clamp5(x[e] + kH*y[e]);
        float ny = clamp5(y[e] + kH*(ai4[e]*in4[e] - gi4[e]*y[e] - wi4[e]*x[e]));
        x[e]=nx; y[e]=ny;
        pk.h[e] = (__bf16)nx;
      }
      *(uint2*)&sA1[a1sIdx] = pk.u;                 // one 8-byte ds_store
    }
    __syncthreads();

    // ---- phase B/C: GEMM1 (dual accumulator chains) + oscillators; waves 0-7 stage half 0 ----
    v16bf a1[4];
#pragma unroll
    for (int kt=0; kt<4; ++kt) a1[kt] = a1v[a1Base + kt*32];
#pragma unroll
    for (int tt=0; tt<8; ++tt){
      v8f ca = zero8, cb = zero8;
      {
        v16bf b0 = gB1[b1Idx + tt*128 +  0];
        v16bf b1 = gB1[b1Idx + tt*128 + 32];
        v16bf b2 = gB1[b1Idx + tt*128 + 64];
        v16bf b3 = gB1[b1Idx + tt*128 + 96];
        ca = wmma_bf16(a1[0], b0, ca);
        cb = wmma_bf16(a1[1], b1, cb);
        ca = wmma_bf16(a1[2], b2, ca);
        cb = wmma_bf16(a1[3], b3, cb);
      }
      const v8f c = ca + cb;
      const float4 pr = pr4[prIdx + tt*16];   // {w2, 2|g|, alpha, b_proj}
      const float4 po = po4[prIdx + tt*16];   // {w2, 2|g|, alpha, 0}
      v8f xr = Xr[tt], xo = Xo[tt];
#pragma unroll
      for (int v=0; v<8; ++v){
        float yr = sYr[yB + v*kYS + tt*16];
        float yo = sYo[yB + v*kYS + tt*16];
        float f   = c[v] + pr.w;
        float ar  = pr.z*f - pr.y*yr - pr.x*xr[v];
        float nxr = clamp5(xr[v] + kH*yr);
        float nyr = clamp5(yr + kH*ar);
        float ao  = po.z*nxr - po.y*yo - po.x*xo[v];
        float nxo = clamp5(xo[v] + kH*yo);
        float nyo = clamp5(yo + kH*ao);
        float d = nxo - nxr; lout += d*d;
        sYr[yB + v*kYS + tt*16] = nyr;
        sYo[yB + v*kYS + tt*16] = nyo;
        xr[v] = nxr; xo[v] = nxo;
      }
      Xr[tt] = xr; Xo[tt] = xo;
      if (w < 8){                               // wave-uniform: this wave owns r < 1024
#pragma unroll
        for (int v=0; v<8; ++v)
          sA2[stBase + (tt>>1)*512 + (tt&1)*8 + v*16] = (__bf16)xo[v];
      }
    }
    __syncthreads();

    // ---- phase D pass 0: readout GEMM over K = 0..1023 (dual chains) ----
    v8f c2a = zero8, c2b = zero8;
#pragma unroll
    for (int cc=0; cc<16; cc+=2){
      v16bf aa = a2v[a2Base + cc*32];
      v16bf ba = gB2[b2Idx + cc*32];
      v16bf ab = a2v[a2Base + (cc+1)*32];
      v16bf bb = gB2[b2Idx + (cc+1)*32];
      c2a = wmma_bf16(aa, ba, c2a);
      c2b = wmma_bf16(ab, bb, c2b);
    }
    __syncthreads();

    // waves 8-15 stage K-half 1 (their x_out columns, r >= 1024)
    if (w >= 8){
#pragma unroll
      for (int tt=0; tt<8; ++tt){
        v8f xo = Xo[tt];
#pragma unroll
        for (int v=0; v<8; ++v)
          sA2[stBase + (tt>>1)*512 + (tt&1)*8 + v*16] = (__bf16)xo[v];
      }
    }
    __syncthreads();

    // ---- phase D pass 1: readout GEMM over K = 1024..2047 ----
#pragma unroll
    for (int cc=0; cc<16; cc+=2){
      v16bf aa = a2v[a2Base + cc*32];
      v16bf ba = gB2[b2Idx + 1024 + cc*32];
      v16bf ab = a2v[a2Base + (cc+1)*32];
      v16bf bb = gB2[b2Idx + 1024 + (cc+1)*32];
      c2a = wmma_bf16(aa, ba, c2a);
      c2b = wmma_bf16(ab, bb, c2b);
    }
    {
      const v8f c2 = c2a + c2b;
#pragma unroll
      for (int v=0; v<8; ++v)
        atomicAdd(&sPred[pBase + v*kD], c2[v]);   // ds_add_f32 (2 contributors)
    }
    __syncthreads();

    // ---- phase E: bias + store preds, re-zero accumulator ----
    {
      const float4 br = *(const float4*)&piB[piIdx];
      float4 pv = *(const float4*)&sPred[elem];
      pv.x += br.x; pv.y += br.y; pv.z += br.z; pv.w += br.w;
      *(float4*)&preds[gbase + (long)t*kD] = pv;
      *(float4*)&sPred[elem] = make_float4(0.f,0.f,0.f,0.f);
    }
    __syncthreads();
  }

  // ---- loss block-reduction -> per-workgroup partials ----
  sPred[tid] = lin;
  sPred[512 + tid] = lout;
  __syncthreads();
  for (int s=256; s>0; s>>=1){
    if (tid < s){
      sPred[tid]       += sPred[tid + s];
      sPred[512 + tid] += sPred[512 + tid + s];
    }
    __syncthreads();
  }
  if (tid == 0){
    partials[blockIdx.x]      = sPred[0];
    partials[16 + blockIdx.x] = sPred[512];
  }
}

__global__ void horn_finalize(const float* __restrict__ partials, float* __restrict__ lossOut){
  if (threadIdx.x == 0 && blockIdx.x == 0){
    float a = 0.f, b = 0.f;
    for (int i=0;i<16;++i){ a += partials[i]; b += partials[16+i]; }
    lossOut[0] = a * (1.f/((float)kB*(float)kD));
    lossOut[1] = b * (1.f/((float)kB*(float)kR));
  }
}

extern "C" void kernel_launch(void* const* d_in, const int* in_sizes, int n_in,
                              void* d_out, int out_size, void* d_ws, size_t ws_size,
                              hipStream_t stream){
  (void)in_sizes; (void)n_in; (void)out_size; (void)ws_size;
  const float* inputs    = (const float*)d_in[0];
  const float* omega_in  = (const float*)d_in[1];
  const float* gamma_in  = (const float*)d_in[2];
  const float* alpha_in  = (const float*)d_in[3];
  const float* W_proj    = (const float*)d_in[4];
  const float* b_proj    = (const float*)d_in[5];
  const float* omega_res = (const float*)d_in[6];
  const float* gamma_res = (const float*)d_in[7];
  const float* alpha_res = (const float*)d_in[8];
  const float* omega_out = (const float*)d_in[9];
  const float* gamma_out = (const float*)d_in[10];
  const float* alpha_out = (const float*)d_in[11];
  const float* W_read    = (const float*)d_in[12];
  const float* b_read    = (const float*)d_in[13];

  char* ws = (char*)d_ws;
  __bf16* f1 = (__bf16*)(ws + kOffF1);
  __bf16* f2 = (__bf16*)(ws + kOffF2);
  float4* pr4 = (float4*)(ws + kOffPR);
  float4* po4 = (float4*)(ws + kOffPO);
  float* piW = (float*)(ws + kOffPIW);
  float* piG = (float*)(ws + kOffPIG);
  float* piA = (float*)(ws + kOffPIA);
  float* piB = (float*)(ws + kOffPIB);
  float* partials = (float*)(ws + kOffPart);

  float* preds = (float*)d_out;
  float* lossOut = preds + (size_t)kB*kS*kD;

  horn_prep<<<dim3(256), dim3(1024), 0, stream>>>(W_proj, W_read,
      omega_in, gamma_in, alpha_in, b_proj,
      omega_res, gamma_res, alpha_res,
      omega_out, gamma_out, alpha_out, b_read,
      f1, f2, pr4, po4, piW, piG, piA, piB);
  horn_main<<<dim3(16), dim3(512), 0, stream>>>(inputs,
      (const v16bf*)f1, (const v16bf*)f2, pr4, po4, piW, piG, piA, piB,
      preds, partials);
  horn_finalize<<<dim3(1), dim3(32), 0, stream>>>(partials, lossOut);
}
